// Packetbnn_7456063226562
// MI455X (gfx1250) — matrix operand order repaired and checked by reading, not compiled, not run on hardware
//
#include <hip/hip_runtime.h>
#include <hip/hip_bf16.h>

#define D 120
#define KP 128
#define LDSW 136   // padded f16 row stride for wb in LDS (272B -> bank-conflict-free b128 loads)
#define EPS 1e-5f

typedef _Float16 half8 __attribute__((ext_vector_type(8)));
typedef _Float16 v16h  __attribute__((ext_vector_type(16)));
typedef float    v8f   __attribute__((ext_vector_type(8)));

__device__ __forceinline__ float binar(float t) {
    // (sign(t)+1)*0.5 : >0 -> 1, <0 -> 0, ==0 (incl -0) -> 0.5
    return t > 0.f ? 1.f : (t < 0.f ? 0.f : 0.5f);
}

// ---------------------------------------------------------------------------
// Kernel 1: h[B,120] = binarize(x)[B,120] @ binarize(conv_w)[120,120]^T
// f16 WMMA, exact for {0,0.5,1} operands with f32 accumulation.
// Block = 256 thr (8 waves). Block owns 128 rows; wave w owns 16-row M-tile.
// wb staged binarized->f16 in LDS, zero-padded to 128x128.
// ---------------------------------------------------------------------------
__global__ void __launch_bounds__(256)
k_gemm(const float* __restrict__ x, const float* __restrict__ w,
       float* __restrict__ h, int B) {
    __shared__ __align__(16) _Float16 lw[128 * LDSW];

    const int tid = threadIdx.x;
    // stage binarized wb (f16), zero pad rows/cols 120..127
    for (int idx = tid; idx < 128 * 128; idx += 256) {
        int n = idx >> 7, k = idx & 127;
        float v = (n < D && k < D) ? binar(w[n * D + k]) : 0.f;
        lw[n * LDSW + k] = (_Float16)v;
    }
    __syncthreads();

    const int wave = tid >> 5;
    const int lane = tid & 31;
    const int r    = lane & 15;          // M (A) / N (B) index within tile
    const int hi   = lane >> 4;          // half-wave select
    const int m0   = blockIdx.x * 128 + wave * 16;
    const int row  = m0 + r;
    const float* xr = x + (long)row * D;

    // Preload 4 A-chunk fragments (K = 0..127), documented 16-bit A layout:
    // lane<16: elements 0..7 -> K=k0..k0+7, 8..15 -> K=k0+16..k0+23 (k0=kc*32)
    // lane>=16: same with +8 offset.
    const int kb = hi * 8;
    v16h a[4];
    for (int kc = 0; kc < 4; ++kc) {
        const int k0 = kc * 32 + kb;
        v16h af;
        float4 p0 = *(const float4*)(xr + k0);
        float4 p1 = *(const float4*)(xr + k0 + 4);
        af[0] = (_Float16)binar(p0.x); af[1] = (_Float16)binar(p0.y);
        af[2] = (_Float16)binar(p0.z); af[3] = (_Float16)binar(p0.w);
        af[4] = (_Float16)binar(p1.x); af[5] = (_Float16)binar(p1.y);
        af[6] = (_Float16)binar(p1.z); af[7] = (_Float16)binar(p1.w);
        if (k0 + 16 < D) {               // only (kc==3, hi==1) run hits the K pad
            float4 q0 = *(const float4*)(xr + k0 + 16);
            float4 q1 = *(const float4*)(xr + k0 + 20);
            af[8]  = (_Float16)binar(q0.x); af[9]  = (_Float16)binar(q0.y);
            af[10] = (_Float16)binar(q0.z); af[11] = (_Float16)binar(q0.w);
            af[12] = (_Float16)binar(q1.x); af[13] = (_Float16)binar(q1.y);
            af[14] = (_Float16)binar(q1.z); af[15] = (_Float16)binar(q1.w);
        } else {
            for (int e = 8; e < 16; ++e) af[e] = (_Float16)0.f;
        }
        a[kc] = af;
    }

    // 8 N-tiles of 16 output features
    for (int nt = 0; nt < 8; ++nt) {
        const int n0 = nt * 16;
        v8f c = {};
        for (int kc = 0; kc < 4; ++kc) {
            // B (KxN) fragment: lane%16 = N; same K layout as A (k0+0..7, k0+16..23, k0 = kc*32 + 8*(lane/16))
            const int k0b = kc * 32 + hi * 8;
            const _Float16* bp = &lw[(n0 + r) * LDSW + k0b];
            half8 blo = *(const half8*)bp;
            half8 bhi = *(const half8*)(bp + 16);
            v16h bf;
            for (int e = 0; e < 8; ++e) { bf[e] = blo[e]; bf[8 + e] = bhi[e]; }
            c = __builtin_amdgcn_wmma_f32_16x16x32_f16(
                    false, a[kc], false, bf, (short)0, c, false, false);
        }
        // C/D layout: vgpr rr, lane l -> M = rr + 8*(l>>4), N = l&15
        const int col = n0 + r;
        if (col < D) {
            for (int rr = 0; rr < 8; ++rr) {
                int orow = m0 + rr + (hi << 3);
                h[(long)orow * D + col] = c[rr];
            }
        }
    }
}

// ---------------------------------------------------------------------------
// Column sum / sumsq over h[B,120] (grid-strided, LDS partials, few atomics)
// ---------------------------------------------------------------------------
__global__ void __launch_bounds__(256)
k_colstats(const float* __restrict__ h, float* __restrict__ sum,
           float* __restrict__ sq, long n) {
    __shared__ float lsum[D], lsq[D];
    for (int t = threadIdx.x; t < D; t += 256) { lsum[t] = 0.f; lsq[t] = 0.f; }
    __syncthreads();
    const long stride = (long)gridDim.x * 256;
    for (long i = blockIdx.x * 256L + threadIdx.x; i < n; i += stride) {
        int col = (int)(i % D);
        float v = h[i];
        atomicAdd(&lsum[col], v);
        atomicAdd(&lsq[col], v * v);
    }
    __syncthreads();
    for (int t = threadIdx.x; t < D; t += 256) {
        atomicAdd(&sum[t], lsum[t]);
        atomicAdd(&sq[t], lsq[t]);
    }
}

// per-feature BN finalize: scale = gamma*rsqrt(var+eps), shift = beta - mean*scale
__global__ void k_finalize(const float* __restrict__ sum, const float* __restrict__ sq,
                           const float* __restrict__ gamma, const float* __restrict__ beta,
                           float* __restrict__ scale, float* __restrict__ shift, float invB) {
    int t = threadIdx.x;
    if (t < D) {
        float m = sum[t] * invB;
        float v = sq[t] * invB - m * m;
        float s = gamma[t] * rsqrtf(v + EPS);
        scale[t] = s;
        shift[t] = beta[t] - m * s;
    }
}

// Fused: g = softsign(BN2(h)) in-place, plus BN1 partial stats of g
__global__ void __launch_bounds__(256)
k_bnsoftsign(float* __restrict__ h, const float* __restrict__ sc,
             const float* __restrict__ sh, float* __restrict__ sum,
             float* __restrict__ sq, long n) {
    __shared__ float lsum[D], lsq[D], lsc[D], lsh[D];
    for (int t = threadIdx.x; t < D; t += 256) {
        lsum[t] = 0.f; lsq[t] = 0.f; lsc[t] = sc[t]; lsh[t] = sh[t];
    }
    __syncthreads();
    const long stride = (long)gridDim.x * 256;
    for (long i = blockIdx.x * 256L + threadIdx.x; i < n; i += stride) {
        int col = (int)(i % D);
        float v = fmaf(h[i], lsc[col], lsh[col]);
        v = v / (1.f + fabsf(v));
        h[i] = v;
        atomicAdd(&lsum[col], v);
        atomicAdd(&lsq[col], v * v);
    }
    __syncthreads();
    for (int t = threadIdx.x; t < D; t += 256) {
        atomicAdd(&sum[t], lsum[t]);
        atomicAdd(&sq[t], lsq[t]);
    }
}

// Wave-per-row: hb = binarize(BN1(g)); o = hb @ binarize(lin_w)^T + lin_b;
// also accumulates final-BN stats (2 classes).
__global__ void __launch_bounds__(256)
k_linear(const float* __restrict__ g, const float* __restrict__ lin_w,
         const float* __restrict__ lin_b, const float* __restrict__ sc,
         const float* __restrict__ sh, float* __restrict__ o,
         float* __restrict__ stat3, int B) {
    __shared__ float w0[D], w1[D], s1[D], b1[D];
    __shared__ float ls[4];
    const int tid = threadIdx.x;
    if (tid < 4) ls[tid] = 0.f;
    for (int t = tid; t < D; t += 256) {
        w0[t] = binar(lin_w[t]);
        w1[t] = binar(lin_w[D + t]);
        s1[t] = sc[t];
        b1[t] = sh[t];
    }
    __syncthreads();
    const float bias0 = lin_b[0], bias1 = lin_b[1];
    const int lane = tid & 31;
    const int gw = blockIdx.x * 8 + (tid >> 5);
    const int nw = gridDim.x * 8;
    for (int row = gw; row < B; row += nw) {
        const float* gr = g + (long)row * D;
        float acc0 = 0.f, acc1 = 0.f;
        for (int k = lane; k < D; k += 32) {
            float hb = binar(fmaf(gr[k], s1[k], b1[k]));
            acc0 += hb * w0[k];
            acc1 += hb * w1[k];
        }
        for (int off = 16; off > 0; off >>= 1) {
            acc0 += __shfl_xor(acc0, off, 32);
            acc1 += __shfl_xor(acc1, off, 32);
        }
        if (lane == 0) {
            float o0 = acc0 + bias0, o1 = acc1 + bias1;
            o[(long)row * 2]     = o0;
            o[(long)row * 2 + 1] = o1;
            atomicAdd(&ls[0], o0);
            atomicAdd(&ls[1], o1);
            atomicAdd(&ls[2], o0 * o0);
            atomicAdd(&ls[3], o1 * o1);
        }
    }
    __syncthreads();
    if (tid < 4) atomicAdd(&stat3[tid], ls[tid]);
}

__global__ void k_finalize3(const float* __restrict__ stat3, float* __restrict__ bn3,
                            float invB) {
    if (threadIdx.x == 0) {
        float m0 = stat3[0] * invB, m1 = stat3[1] * invB;
        float v0 = stat3[2] * invB - m0 * m0;
        float v1 = stat3[3] * invB - m1 * m1;
        bn3[0] = m0; bn3[1] = m1;
        bn3[2] = rsqrtf(v0 + EPS); bn3[3] = rsqrtf(v1 + EPS);
    }
}

__global__ void __launch_bounds__(256)
k_logsoftmax(const float* __restrict__ o, const float* __restrict__ bn3,
             float* __restrict__ out, int B) {
    const float m0 = bn3[0], m1 = bn3[1], r0 = bn3[2], r1 = bn3[3];
    const long stride = (long)gridDim.x * 256;
    for (long row = blockIdx.x * 256L + threadIdx.x; row < B; row += stride) {
        float a0 = (o[row * 2]     - m0) * r0;
        float a1 = (o[row * 2 + 1] - m1) * r1;
        float mx = fmaxf(a0, a1);
        float l  = mx + logf(expf(a0 - mx) + expf(a1 - mx));
        out[row * 2]     = a0 - l;
        out[row * 2 + 1] = a1 - l;
    }
}

__global__ void k_zero(float* __restrict__ p, int n) {
    int i = blockIdx.x * 256 + threadIdx.x;
    if (i < n) p[i] = 0.f;
}

// ---------------------------------------------------------------------------
extern "C" void kernel_launch(void* const* d_in, const int* in_sizes, int n_in,
                              void* d_out, int out_size, void* d_ws, size_t ws_size,
                              hipStream_t stream) {
    const float* x      = (const float*)d_in[0];
    const float* conv_w = (const float*)d_in[1];
    const float* g2     = (const float*)d_in[2];
    const float* b2     = (const float*)d_in[3];
    const float* g1     = (const float*)d_in[4];
    const float* b1     = (const float*)d_in[5];
    const float* lw     = (const float*)d_in[6];
    const float* lb     = (const float*)d_in[7];
    float* out = (float*)d_out;

    const int B = in_sizes[0] / D;   // 262144
    const float invB = 1.f / (float)B;

    float* ws   = (float*)d_ws;
    float* h    = ws;                        // B*120
    float* o    = h + (long)B * D;           // B*2
    float* st   = o + (long)B * 2;           // stats region
    float* s2sum = st;        float* s2sq = st + 120;
    float* sc2   = st + 240;  float* sh2  = st + 360;
    float* s1sum = st + 480;  float* s1sq = st + 600;
    float* sc1   = st + 720;  float* sh1  = st + 840;
    float* stat3 = st + 960;  float* bn3  = st + 964;

    k_zero<<<4, 256, 0, stream>>>(st, 968);
    k_gemm<<<B / 128, 256, 0, stream>>>(x, conv_w, h, B);
    k_colstats<<<512, 256, 0, stream>>>(h, s2sum, s2sq, (long)B * D);
    k_finalize<<<1, 128, 0, stream>>>(s2sum, s2sq, g2, b2, sc2, sh2, invB);
    k_bnsoftsign<<<512, 256, 0, stream>>>(h, sc2, sh2, s1sum, s1sq, (long)B * D);
    k_finalize<<<1, 128, 0, stream>>>(s1sum, s1sq, g1, b1, sc1, sh1, invB);
    k_linear<<<1024, 256, 0, stream>>>(h, lw, lb, sc1, sh1, o, stat3, B);
    k_finalize3<<<1, 64, 0, stream>>>(stat3, bn3, invB);
    k_logsoftmax<<<512, 256, 0, stream>>>(o, bn3, out, B);
}
